// MiMoV2Attention_30133490548818
// MI455X (gfx1250) — matrix operand order repaired and checked by reading, not compiled, not run on hardware
//
#include <hip/hip_runtime.h>
#include <hip/hip_bf16.h>

// ---- problem constants (match reference) ----
constexpr int SEQ  = 2048;
constexpr int HIDN = 2048;
constexpr int NH   = 32;
constexpr int KVHN = 8;
constexpr int HD   = 64;
constexpr int WINS = 512;
constexpr int GQA  = NH / KVHN;                 // 4
constexpr int NQKV = NH*HD + 2*KVHN*HD;         // 3072

typedef __attribute__((ext_vector_type(16))) __bf16 v16bf;
typedef __attribute__((ext_vector_type(8)))  __bf16 v8bf;
typedef __attribute__((ext_vector_type(8)))  float  v8f;

static __device__ __forceinline__ v16bf cat8(v8bf lo, v8bf hi) {
  return __builtin_shufflevector(lo, hi, 0,1,2,3,4,5,6,7,8,9,10,11,12,13,14,15);
}

#define WMMA_BF16(a,b,c) \
  __builtin_amdgcn_wmma_f32_16x16x32_bf16(false,(a),false,(b),(short)0,(c),false,false)

// ---------------------------------------------------------------------------
// 1) f32 -> bf16 elementwise
// ---------------------------------------------------------------------------
__global__ void f32_to_bf16(const float* __restrict__ x, __bf16* __restrict__ y, int n) {
  int i = blockIdx.x * blockDim.x + threadIdx.x;
  if (i < n) y[i] = (__bf16)x[i];
}

// ---------------------------------------------------------------------------
// 2) W[K][N] f32  ->  Wt[N][K] bf16  (K-contiguous for WMMA B-fragments)
// ---------------------------------------------------------------------------
__global__ void transpose_to_bf16(const float* __restrict__ W, __bf16* __restrict__ Wt,
                                  int K, int N) {
  int i = blockIdx.x * blockDim.x + threadIdx.x;   // linear over N*K (Wt layout)
  if (i >= N * K) return;
  int n = i / K;
  int k = i - n * K;
  Wt[i] = (__bf16)W[(size_t)k * N + n];
}

// ---------------------------------------------------------------------------
// 3) GEMM: C[M][N] f32 = A[M][K] bf16 @ Bt[N][K] bf16   (dims mult of 64/128)
//    One wave computes a 32x64 C tile: 2 M-frags x 4 N-frags, K-step 32.
// ---------------------------------------------------------------------------
__global__ __launch_bounds__(128) void gemm_bf16(const __bf16* __restrict__ A,
                                                 const __bf16* __restrict__ Bt,
                                                 float* __restrict__ C,
                                                 int M, int N, int K) {
  const int lane = threadIdx.x & 31;
  const int wave = threadIdx.x >> 5;
  const int m0 = blockIdx.y * 64  + (wave >> 1) * 32;
  const int n0 = blockIdx.x * 128 + (wave & 1) * 64;
  const int mr = lane & 15;
  const bool lo16 = lane < 16;
  const int ab = lo16 ? 0 : 16;    // A-frag K base (contiguous k = 16h + s, matches B)
  const int bb = lo16 ? 0 : 16;    // B-frag K base (contiguous k = 16h + s, matches A)

  const __bf16* a0p = A + (size_t)(m0 + mr) * K;
  const __bf16* a1p = A + (size_t)(m0 + 16 + mr) * K;
  const __bf16* b0p = Bt + (size_t)(n0 +  0 + mr) * K + bb;
  const __bf16* b1p = Bt + (size_t)(n0 + 16 + mr) * K + bb;
  const __bf16* b2p = Bt + (size_t)(n0 + 32 + mr) * K + bb;
  const __bf16* b3p = Bt + (size_t)(n0 + 48 + mr) * K + bb;

  v8f acc[2][4] = {};
  for (int k0 = 0; k0 < K; k0 += 32) {
    __builtin_prefetch(a0p + k0 + 256, 0, 0);
    __builtin_prefetch(b0p + k0 + 256, 0, 0);
    v16bf a0 = cat8(*(const v8bf*)(a0p + k0 + ab), *(const v8bf*)(a0p + k0 + ab + 8));
    v16bf a1 = cat8(*(const v8bf*)(a1p + k0 + ab), *(const v8bf*)(a1p + k0 + ab + 8));
    v16bf b0 = *(const v16bf*)(b0p + k0);
    v16bf b1 = *(const v16bf*)(b1p + k0);
    v16bf b2 = *(const v16bf*)(b2p + k0);
    v16bf b3 = *(const v16bf*)(b3p + k0);
    acc[0][0] = WMMA_BF16(a0, b0, acc[0][0]);
    acc[0][1] = WMMA_BF16(a0, b1, acc[0][1]);
    acc[0][2] = WMMA_BF16(a0, b2, acc[0][2]);
    acc[0][3] = WMMA_BF16(a0, b3, acc[0][3]);
    acc[1][0] = WMMA_BF16(a1, b0, acc[1][0]);
    acc[1][1] = WMMA_BF16(a1, b1, acc[1][1]);
    acc[1][2] = WMMA_BF16(a1, b2, acc[1][2]);
    acc[1][3] = WMMA_BF16(a1, b3, acc[1][3]);
  }

  const int rb = lo16 ? 0 : 8;     // C-frag row base (ISA 32-bit C layout)
#pragma unroll
  for (int i = 0; i < 2; ++i)
#pragma unroll
    for (int f = 0; f < 4; ++f)
#pragma unroll
      for (int r = 0; r < 8; ++r)
        C[(size_t)(m0 + i*16 + r + rb) * N + (n0 + f*16 + mr)] = acc[i][f][r];
}

// ---------------------------------------------------------------------------
// 4) RoPE (NeoX) + scatter:
//    qkv f32 [SEQ][3072] -> Qbf [SEQ][NH][HD] (pre-scaled by D^-0.5),
//    Kbf [SEQ][KVHN][HD], Vt bf16 [KVHN*HD][SEQ] (transposed), kv_fused f32.
// ---------------------------------------------------------------------------
__global__ __launch_bounds__(256) void rope_scatter(const int* __restrict__ pos,
                                                    const float* __restrict__ qkv,
                                                    __bf16* __restrict__ qbf,
                                                    __bf16* __restrict__ kbf,
                                                    __bf16* __restrict__ vtb,
                                                    float* __restrict__ kvf) {
  const int s = blockIdx.x;
  const float p = (float)pos[s];
  const float* row = qkv + (size_t)s * NQKV;
  const float lnt_over_half = 13.815510557964274f / 32.0f;  // ln(1e6)/(D/2)

  for (int idx = threadIdx.x; idx < (NH + KVHN) * 32; idx += blockDim.x) {
    int head = idx >> 5, i = idx & 31;
    float invf = __expf(-(float)i * lnt_over_half);
    float ang = p * invf;
    float sn, cs;
    __sincosf(ang, &sn, &cs);
    if (head < NH) {
      const float* b = row + head * HD;
      float x1 = b[i], x2 = b[i + 32];
      float o1 = x1 * cs - x2 * sn;
      float o2 = x2 * cs + x1 * sn;
      __bf16* q = qbf + ((size_t)s * NH + head) * HD;
      q[i]      = (__bf16)(o1 * 0.125f);       // fold in D^-0.5
      q[i + 32] = (__bf16)(o2 * 0.125f);
    } else {
      int kh = head - NH;
      const float* b = row + NH * HD + kh * HD;
      float x1 = b[i], x2 = b[i + 32];
      float o1 = x1 * cs - x2 * sn;
      float o2 = x2 * cs + x1 * sn;
      __bf16* kk = kbf + ((size_t)s * KVHN + kh) * HD;
      kk[i]      = (__bf16)o1;
      kk[i + 32] = (__bf16)o2;
      float* kv = kvf + (size_t)s * (2 * KVHN * HD);
      kv[kh * HD + i]      = o1;
      kv[kh * HD + i + 32] = o2;
    }
  }
  for (int idx = threadIdx.x; idx < KVHN * HD; idx += blockDim.x) {
    float v = row[NH * HD + KVHN * HD + idx];
    kvf[(size_t)s * (2 * KVHN * HD) + KVHN * HD + idx] = v;
    vtb[(size_t)idx * SEQ + s] = (__bf16)v;    // transposed for PV B-fragments
  }
}

// ---------------------------------------------------------------------------
// 5) Windowed flash attention with sink. One wave per (head, 16-query tile).
//    32-key blocks: 4 WMMAs Q@K^T + 4 WMMAs P@V; online softmax, m0=sink,l0=1.
// ---------------------------------------------------------------------------
__global__ __launch_bounds__(128) void attn_fwd(const __bf16* __restrict__ Qbf,
                                                const __bf16* __restrict__ Kbf,
                                                const __bf16* __restrict__ Vt,
                                                const float* __restrict__ sink,
                                                __bf16* __restrict__ Obf) {
  __shared__ __bf16 pls[4][16 * 32];           // per-wave P staging (C->A relayout)
  const int lane = threadIdx.x & 31;
  const int wave = threadIdx.x >> 5;
  const int h   = blockIdx.y * 4 + wave;
  const int q0  = blockIdx.x * 16;
  const int kvh = h / GQA;
  const int mr  = lane & 15;
  const bool lo16 = lane < 16;
  const int ab = lo16 ? 0 : 16;                // A-frag K base
  const int rb = lo16 ? 0 : 8;                 // C-frag row base

  // Q A-fragments (row q0+mr per lane; d-chunks 0..31 and 32..63)
  const __bf16* qrow = Qbf + ((size_t)(q0 + mr) * NH + h) * HD;
  v16bf qa0 = cat8(*(const v8bf*)(qrow +  0 + ab), *(const v8bf*)(qrow +  0 + ab + 8));
  v16bf qa1 = cat8(*(const v8bf*)(qrow + 32 + ab), *(const v8bf*)(qrow + 32 + ab + 8));

  const float sk = sink[h];
  float mrun[8], lrun[8];
#pragma unroll
  for (int r = 0; r < 8; ++r) { mrun[r] = sk; lrun[r] = 1.0f; }
  v8f o[4] = {};

  int jstart = q0 - (WINS - 1);
  if (jstart < 0) jstart = 0;
  jstart &= ~31;                                // 32-aligned key blocks

  __bf16* pw = &pls[wave][0];

  for (int j0 = jstart; j0 < q0 + 16; j0 += 32) {
    // ---- scores: S = Q @ K^T (K rows are d-contiguous -> direct B-frags) ----
    const __bf16* kp0 = Kbf + ((size_t)(j0 + mr) * KVHN + kvh) * HD + (lo16 ? 0 : 16);
    const __bf16* kp1 = Kbf + ((size_t)(j0 + 16 + mr) * KVHN + kvh) * HD + (lo16 ? 0 : 16);
    v16bf b00 = *(const v16bf*)(kp0);
    v16bf b01 = *(const v16bf*)(kp0 + 32);
    v16bf b10 = *(const v16bf*)(kp1);
    v16bf b11 = *(const v16bf*)(kp1 + 32);
    v8f s0 = {}, s1 = {};
    s0 = WMMA_BF16(qa0, b00, s0);
    s0 = WMMA_BF16(qa1, b01, s0);
    s1 = WMMA_BF16(qa0, b10, s1);
    s1 = WMMA_BF16(qa1, b11, s1);

    // ---- mask + online softmax (row stats live in each 16-lane half) ----
#pragma unroll
    for (int r = 0; r < 8; ++r) {
      const int i = q0 + r + rb;
      const int jc0 = j0 + mr, jc1 = jc0 + 16;
      float v0 = s0[r], v1 = s1[r];
      if (jc0 > i || (i - jc0) >= WINS) v0 = -1e30f;
      if (jc1 > i || (i - jc1) >= WINS) v1 = -1e30f;
      float rmax = fmaxf(v0, v1);
      rmax = fmaxf(rmax, __shfl_xor(rmax, 1));
      rmax = fmaxf(rmax, __shfl_xor(rmax, 2));
      rmax = fmaxf(rmax, __shfl_xor(rmax, 4));
      rmax = fmaxf(rmax, __shfl_xor(rmax, 8));
      const float mnew  = fmaxf(mrun[r], rmax);
      const float scale = __expf(mrun[r] - mnew);
      const float p0 = __expf(v0 - mnew);
      const float p1 = __expf(v1 - mnew);
      float rsum = p0 + p1;
      rsum += __shfl_xor(rsum, 1);
      rsum += __shfl_xor(rsum, 2);
      rsum += __shfl_xor(rsum, 4);
      rsum += __shfl_xor(rsum, 8);
      lrun[r] = lrun[r] * scale + rsum;
      mrun[r] = mnew;
#pragma unroll
      for (int f = 0; f < 4; ++f) o[f][r] *= scale;
      const int prow = r + rb;
      pw[prow * 32 + mr]      = (__bf16)p0;    // C-layout -> LDS
      pw[prow * 32 + 16 + mr] = (__bf16)p1;
    }

    // LDS is in-order within a wave: read back in A-layout
    v16bf pa = cat8(*(const v8bf*)(pw + mr * 32 + ab),
                    *(const v8bf*)(pw + mr * 32 + ab + 8));

    // ---- P @ V  (V transposed [d][s] -> contiguous B-frags) ----
#pragma unroll
    for (int f = 0; f < 4; ++f) {
      const __bf16* vp = Vt + ((size_t)(kvh * HD + f * 16 + mr)) * SEQ + j0 + (lo16 ? 0 : 16);
      v16bf bv = *(const v16bf*)vp;
      o[f] = WMMA_BF16(pa, bv, o[f]);
    }
  }

  // ---- normalize + store attn_out bf16 [SEQ][NH*HD] ----
#pragma unroll
  for (int f = 0; f < 4; ++f)
#pragma unroll
    for (int r = 0; r < 8; ++r) {
      float val = o[f][r] / lrun[r];
      Obf[(size_t)(q0 + r + rb) * (NH * HD) + h * HD + f * 16 + mr] = (__bf16)val;
    }
}

// ---------------------------------------------------------------------------
extern "C" void kernel_launch(void* const* d_in, const int* in_sizes, int n_in,
                              void* d_out, int out_size, void* d_ws, size_t ws_size,
                              hipStream_t stream) {
  const int*   positions = (const int*)d_in[0];
  const float* hs   = (const float*)d_in[1];
  const float* wq   = (const float*)d_in[2];
  const float* wk   = (const float*)d_in[3];
  const float* wv   = (const float*)d_in[4];
  const float* wo   = (const float*)d_in[5];
  const float* sink = (const float*)d_in[6];

  float* out = (float*)d_out;                        // [SEQ][HIDN]
  float* kvf = out + (size_t)SEQ * HIDN;             // [SEQ][2*KVHN*HD]

  // workspace carve-up
  __bf16* hsb = (__bf16*)d_ws;                       // SEQ*HIDN
  __bf16* btq = hsb + (size_t)SEQ * HIDN;            // NQKV*HIDN  (W_qkv^T)
  __bf16* wto = btq + (size_t)NQKV * HIDN;           // HIDN*HIDN  (W_o^T)
  float*  qkv = (float*)(wto + (size_t)HIDN * HIDN); // SEQ*NQKV
  __bf16* qbf = (__bf16*)(qkv + (size_t)SEQ * NQKV); // SEQ*NH*HD
  __bf16* kbf = qbf + (size_t)SEQ * NH * HD;         // SEQ*KVHN*HD
  __bf16* vtb = kbf + (size_t)SEQ * KVHN * HD;       // KVHN*HD x SEQ (transposed)
  __bf16* obf = vtb + (size_t)SEQ * KVHN * HD;       // SEQ*NH*HD

  // 1) convert activations
  f32_to_bf16<<<(SEQ * HIDN + 255) / 256, 256, 0, stream>>>(hs, hsb, SEQ * HIDN);

  // 2) transpose+convert weights (packed QKV B^T, then W_o^T)
  transpose_to_bf16<<<(HIDN * NH * HD + 255) / 256, 256, 0, stream>>>(
      wq, btq, HIDN, NH * HD);
  transpose_to_bf16<<<(HIDN * KVHN * HD + 255) / 256, 256, 0, stream>>>(
      wk, btq + (size_t)NH * HD * HIDN, HIDN, KVHN * HD);
  transpose_to_bf16<<<(HIDN * KVHN * HD + 255) / 256, 256, 0, stream>>>(
      wv, btq + (size_t)(NH * HD + KVHN * HD) * HIDN, HIDN, KVHN * HD);
  transpose_to_bf16<<<(HIDN * HIDN + 255) / 256, 256, 0, stream>>>(
      wo, wto, HIDN, HIDN);

  // 3) fused QKV projection
  {
    dim3 grid(NQKV / 128, SEQ / 64);
    gemm_bf16<<<grid, 128, 0, stream>>>(hsb, btq, qkv, SEQ, NQKV, HIDN);
  }

  // 4) RoPE + layout scatter + kv_fused output
  rope_scatter<<<SEQ, 256, 0, stream>>>(positions, qkv, qbf, kbf, vtb, kvf);

  // 5) windowed attention with sink
  {
    dim3 grid(SEQ / 16, NH / 4);
    attn_fwd<<<grid, 128, 0, stream>>>(qbf, kbf, vtb, sink, obf);
  }

  // 6) output projection -> d_out
  {
    dim3 grid(HIDN / 128, SEQ / 64);
    gemm_bf16<<<grid, 128, 0, stream>>>(obf, wto, out, SEQ, HIDN, NH * HD);
  }
}
